// HEmbedding_9637906612293
// MI455X (gfx1250) — compile-verified, matrix operand order checked by reading, not run on hardware
//
#include <hip/hip_runtime.h>
#include <hip/hip_bf16.h>
#include <math.h>

// ---------------- problem constants (match reference) ----------------
constexpr int kMC   = 2048;
constexpr int kD    = 128;
constexpr int kB    = 64;
constexpr int kL    = 16;
constexpr int kNOBJ = 128;
constexpr int kFDIM = 512;
constexpr int kNN   = kMC + kNOBJ;   // 2176
constexpr float kTEMP = 3.0f;
constexpr float kTH   = 0.8f;
constexpr float kPOS  = 10.0f;
constexpr float kNEG  = -10.0f;

typedef __attribute__((ext_vector_type(16))) _Float16 v16h;
typedef __attribute__((ext_vector_type(8)))  float    v8f;

// =====================================================================
// Kernel 1: objects = scene @ W_feat + b_feat  via v_wmma_f32_16x16x32_f16
//   M = B*NOBJ = 8192, K = FDIM = 512, N = D = 128
//   One wave (32 lanes) per 16x16 output tile; 16 WMMAs over K.
// =====================================================================
__global__ __launch_bounds__(32)
void objects_wmma_kernel(const float* __restrict__ scene,   // [8192, 512]
                         const float* __restrict__ W,       // [512, 128]
                         const float* __restrict__ bfeat,   // [128]
                         float* __restrict__ obj)           // [8192, 128]
{
    const int tm   = blockIdx.x;            // 0..511  (M tiles)
    const int tn   = blockIdx.y;            // 0..7    (N tiles)
    const int lane = threadIdx.x;           // 0..31
    const int halfK = (lane & 16) ? 8 : 0;  // K-striping per ISA A/B layout
    const int mrow = tm * 16 + (lane & 15); // A row owned by this lane
    const int ncol = tn * 16 + (lane & 15); // B column owned by this lane

    const float* arow = scene + (size_t)mrow * kFDIM;
    const float* bcol = W + ncol;

    v8f acc = {};
    for (int k0 = 0; k0 < kFDIM; k0 += 32) {
        v16h a, bm;
#pragma unroll
        for (int h = 0; h < 8; ++h) {
            // A 16x32 f16 layout: lanes<16 hold K {0..7,16..23}, lanes>=16 hold {8..15,24..31}
            a[h]     = (_Float16)arow[k0 + halfK + h];
            a[h + 8] = (_Float16)arow[k0 + halfK + 16 + h];
            // B 32x16 f16: mirror K striping, lane owns column ncol (stride D)
            bm[h]     = (_Float16)bcol[(size_t)(k0 + halfK + h) * kD];
            bm[h + 8] = (_Float16)bcol[(size_t)(k0 + halfK + 16 + h) * kD];
        }
        acc = __builtin_amdgcn_wmma_f32_16x16x32_f16(
            /*neg_a=*/false, a, /*neg_b=*/false, bm,
            /*c_mod=*/(short)0, acc, /*reuse_a=*/false, /*reuse_b=*/false);
    }

    const float bias  = bfeat[ncol];
    const int   mbase = tm * 16 + ((lane & 16) ? 8 : 0); // C/D: VGPR r -> M = r (+8 for hi lanes)
#pragma unroll
    for (int r = 0; r < 8; ++r) {
        obj[(size_t)(mbase + r) * kD + ncol] = acc[r] + bias;
    }
}

// =====================================================================
// Kernel 2: L2 norms of all node rows.
//   rows 0..MC-1        -> concept_table rows  -> norm_c
//   rows MC..MC+8191    -> objects rows        -> norm_o
//   One wave per row: lane reads float4, shuffle-reduce (wave32).
// =====================================================================
__global__ __launch_bounds__(256)
void norms_kernel(const float* __restrict__ ct,   // [MC, D]
                  const float* __restrict__ obj,  // [B*NOBJ, D]
                  float* __restrict__ norm_c,     // [MC]
                  float* __restrict__ norm_o)     // [B*NOBJ]
{
    const int row  = blockIdx.x * 8 + (threadIdx.x >> 5);
    const int lane = threadIdx.x & 31;
    const int total = kMC + kB * kNOBJ;
    if (row >= total) return;

    const float* p = (row < kMC) ? (ct + (size_t)row * kD)
                                 : (obj + (size_t)(row - kMC) * kD);
    float4 v = *(const float4*)(p + lane * 4);
    float s = v.x * v.x + v.y * v.y + v.z * v.z + v.w * v.w;
#pragma unroll
    for (int m = 16; m > 0; m >>= 1) s += __shfl_xor(s, m, 32);
    if (lane == 0) {
        float r = sqrtf(s);
        if (row < kMC) norm_c[row] = r;
        else           norm_o[row - kMC] = r;
    }
}

// =====================================================================
// Kernel 3: per-batch sequential program executor.
//   One block (256 threads = 8 wave32) per batch. att lives in LDS.
//   Deterministic: per-wave LDS partials + tree sums (no float atomics).
// =====================================================================
__global__ __launch_bounds__(256)
void exec_kernel(const float* __restrict__ concept_table,   // [MC, D]
                 const float* __restrict__ relation_table,  // [MC, D*D]
                 const int*   __restrict__ program,         // [B, L, 2]
                 const float* __restrict__ objects,         // [B*NOBJ, D]
                 const float* __restrict__ norm_c,          // [MC]
                 const float* __restrict__ norm_o,          // [B*NOBJ]
                 float* __restrict__ out,                   // [B, MC]
                 float* __restrict__ hist)                  // [B, L, NN]
{
    __shared__ __align__(16) float att[kNN];
    __shared__ __align__(16) float vec[kD];      // cvec or trans
    __shared__ __align__(16) float gvec[kD];     // gather vector
    __shared__ __align__(16) float gpart[8][kD]; // per-wave partials
    __shared__ float red[256];

    const int b    = blockIdx.x;
    const int t    = threadIdx.x;
    const int lane = t & 31;
    const int wave = t >> 5;    // 0..7

    const float* obj_b   = objects + (size_t)b * kNOBJ * kD;
    const float* norm_ob = norm_o + (size_t)b * kNOBJ;

    for (int n = t; n < kNN; n += 256) att[n] = 1.0f;
    __syncthreads();

    for (int l = 0; l < kL; ++l) {
        const int op  = program[((size_t)b * kL + l) * 2 + 0];
        const int arg = program[((size_t)b * kL + l) * 2 + 1];

        if (op == 1) {                        // sel_obj: zero concept nodes
            for (int n = t; n < kMC; n += 256) att[n] = 0.0f;
            __syncthreads();
        } else if (op == 2) {                 // sel_con: zero object nodes
            for (int n = kMC + t; n < kNN; n += 256) att[n] = 0.0f;
            __syncthreads();
        } else if (op == 3) {                 // verify
            if (t < kD) vec[t] = concept_table[(size_t)arg * kD + t];
            __syncthreads();
            // ||cvec||
            float p = (t < kD) ? vec[t] * vec[t] : 0.0f;
            red[t] = p; __syncthreads();
            for (int s = 128; s > 0; s >>= 1) { if (t < s) red[t] += red[t + s]; __syncthreads(); }
            const float yn = fmaxf(sqrtf(red[0]), 1e-8f);
            __syncthreads();
            // per-node cosine-sim sweep: one wave per node, lanes split D
            for (int n = wave; n < kNN; n += 8) {
                const float* row = (n < kMC) ? (concept_table + (size_t)n * kD)
                                             : (obj_b + (size_t)(n - kMC) * kD);
                float4 v  = *(const float4*)(row + lane * 4);
                float4 c4 = *(const float4*)(vec + lane * 4);
                float d = v.x * c4.x + v.y * c4.y + v.z * c4.z + v.w * c4.w;
#pragma unroll
                for (int m = 16; m > 0; m >>= 1) d += __shfl_xor(d, m, 32);
                if (lane == 0) {
                    float xn  = fmaxf((n < kMC) ? norm_c[n] : norm_ob[n - kMC], 1e-8f);
                    float sim = d / (xn * yn);
                    float sg  = 1.0f / (1.0f + __expf(-(sim - kTH) * kTEMP));
                    att[n] = fminf(att[n], sg);
                }
            }
            __syncthreads();
        } else if (op == 4) {                 // choose
            for (int n = t; n < kMC; n += 256) att[n] = 0.0f;
            __syncthreads();
            if (t == 0) att[arg] = 1.0f;
            __syncthreads();
        } else if (op == 5) {                 // exist
            float m = -INFINITY;
            for (int n = t; n < kNN; n += 256) m = fmaxf(m, att[n]);
            red[t] = m; __syncthreads();
            for (int s = 128; s > 0; s >>= 1) { if (t < s) red[t] = fmaxf(red[t], red[t + s]); __syncthreads(); }
            const float smax = red[0];
            __syncthreads();
            for (int n = t; n < kNN; n += 256) att[n] = kNEG;
            __syncthreads();
            if (t == 0) { att[0] = smax * kPOS; att[1] = kPOS - smax * kPOS; }
            __syncthreads();
        } else if (op == 6) {                 // transfer
            // amax = att.max()
            float m = -INFINITY;
            for (int n = t; n < kNN; n += 256) m = fmaxf(m, att[n]);
            red[t] = m; __syncthreads();
            for (int s = 128; s > 0; s >>= 1) { if (t < s) red[t] = fmaxf(red[t], red[t + s]); __syncthreads(); }
            const float amax = red[0];
            __syncthreads();
            // gather = att @ nodes: wave strides nodes, lanes split D (4 each)
            {
                float4 acc = {0.f, 0.f, 0.f, 0.f};
                for (int n = wave; n < kNN; n += 8) {
                    float a = att[n];
                    const float* row = (n < kMC) ? (concept_table + (size_t)n * kD)
                                                 : (obj_b + (size_t)(n - kMC) * kD);
                    float4 v = *(const float4*)(row + lane * 4);
                    acc.x += a * v.x; acc.y += a * v.y; acc.z += a * v.z; acc.w += a * v.w;
                }
                *(float4*)(&gpart[wave][lane * 4]) = acc;
            }
            __syncthreads();
            if (t < kD) {                     // deterministic cross-wave sum
                float s = 0.0f;
#pragma unroll
                for (int w = 0; w < 8; ++w) s += gpart[w][t];
                gvec[t] = s;
            }
            __syncthreads();
            // trans = gather @ rmat: wave strides rows d, lanes split d'
            {
                const float* R = relation_table + (size_t)arg * (kD * kD);
                float4 acc = {0.f, 0.f, 0.f, 0.f};
                for (int d = wave; d < kD; d += 8) {
                    float gc = gvec[d];
                    float4 v = *(const float4*)(R + (size_t)d * kD + lane * 4);
                    acc.x += gc * v.x; acc.y += gc * v.y; acc.z += gc * v.z; acc.w += gc * v.w;
                }
                *(float4*)(&gpart[wave][lane * 4]) = acc;
            }
            __syncthreads();
            if (t < kD) {
                float s = 0.0f;
#pragma unroll
                for (int w = 0; w < 8; ++w) s += gpart[w][t];
                vec[t] = s;
            }
            __syncthreads();
            // ||trans||
            float p = (t < kD) ? vec[t] * vec[t] : 0.0f;
            red[t] = p; __syncthreads();
            for (int s = 128; s > 0; s >>= 1) { if (t < s) red[t] += red[t + s]; __syncthreads(); }
            const float yn = fmaxf(sqrtf(red[0]), 1e-8f);
            __syncthreads();
            // att = sigmoid((cos(nodes,trans)-TH)*TEMP) * amax
            for (int n = wave; n < kNN; n += 8) {
                const float* row = (n < kMC) ? (concept_table + (size_t)n * kD)
                                             : (obj_b + (size_t)(n - kMC) * kD);
                float4 v  = *(const float4*)(row + lane * 4);
                float4 c4 = *(const float4*)(vec + lane * 4);
                float d = v.x * c4.x + v.y * c4.y + v.z * c4.z + v.w * c4.w;
#pragma unroll
                for (int m = 16; m > 0; m >>= 1) d += __shfl_xor(d, m, 32);
                if (lane == 0) {
                    float xn  = fmaxf((n < kMC) ? norm_c[n] : norm_ob[n - kMC], 1e-8f);
                    float sim = d / (xn * yn);
                    float sg  = 1.0f / (1.0f + __expf(-(sim - kTH) * kTEMP));
                    att[n] = sg * amax;
                }
            }
            __syncthreads();
        }
        // op == 0 (null): fall through

        // record hist[b, l, :]
        float* h = hist + ((size_t)b * kL + l) * kNN;
        for (int n = t; n < kNN; n += 256) h[n] = att[n];
        __syncthreads();
    }

    // out[b, :] = log_softmax(att[:MC])
    float m = -INFINITY;
    for (int n = t; n < kMC; n += 256) m = fmaxf(m, att[n]);
    red[t] = m; __syncthreads();
    for (int s = 128; s > 0; s >>= 1) { if (t < s) red[t] = fmaxf(red[t], red[t + s]); __syncthreads(); }
    const float mx = red[0];
    __syncthreads();
    float se = 0.0f;
    for (int n = t; n < kMC; n += 256) se += __expf(att[n] - mx);
    red[t] = se; __syncthreads();
    for (int s = 128; s > 0; s >>= 1) { if (t < s) red[t] += red[t + s]; __syncthreads(); }
    const float lse = __logf(red[0]);
    for (int n = t; n < kMC; n += 256)
        out[(size_t)b * kMC + n] = att[n] - mx - lse;
}

// =====================================================================
// Host launcher
// =====================================================================
extern "C" void kernel_launch(void* const* d_in, const int* in_sizes, int n_in,
                              void* d_out, int out_size, void* d_ws, size_t ws_size,
                              hipStream_t stream) {
    (void)in_sizes; (void)n_in; (void)out_size; (void)ws_size;

    const float* scene          = (const float*)d_in[0];   // [B, NOBJ, FDIM]
    const int*   program        = (const int*)  d_in[1];   // [B, L, 2]
    const float* concept_table  = (const float*)d_in[2];   // [MC, D]
    const float* relation_table = (const float*)d_in[3];   // [MC, D*D]
    const float* W_feat         = (const float*)d_in[4];   // [FDIM, D]
    const float* b_feat         = (const float*)d_in[5];   // [D]

    float* out  = (float*)d_out;                           // [B, MC]
    float* hist = out + (size_t)kB * kMC;                  // [B, L, NN]

    float* objects = (float*)d_ws;                         // [B*NOBJ, D] = 4 MB
    float* norm_c  = objects + (size_t)kB * kNOBJ * kD;    // [MC]
    float* norm_o  = norm_c + kMC;                         // [B*NOBJ]

    // 1) objects GEMM (WMMA f16 -> f32)
    dim3 g1((kB * kNOBJ) / 16, kD / 16);                   // 512 x 8 tiles
    objects_wmma_kernel<<<g1, 32, 0, stream>>>(scene, W_feat, b_feat, objects);

    // 2) node norms (concepts + per-batch objects)
    const int rows = kMC + kB * kNOBJ;                     // 10240
    norms_kernel<<<(rows + 7) / 8, 256, 0, stream>>>(concept_table, objects, norm_c, norm_o);

    // 3) sequential program execution, one workgroup per batch
    exec_kernel<<<kB, 256, 0, stream>>>(concept_table, relation_table, program,
                                        objects, norm_c, norm_o, out, hist);
}
